// SemanticTextureGatedMamba_70360154243318
// MI455X (gfx1250) — matrix operand order repaired and checked by reading, not compiled, not run on hardware
//
#include <hip/hip_runtime.h>
#include <hip/hip_bf16.h>

// ---------------------------------------------------------------------------
// SemanticTextureGatedMamba for gfx1250 (MI455X, wave32, WMMA + async-to-LDS)
// B=8, L=4096, C=256, D_MODEL=128, D_STATE=16, D_CONV=4, D_INNER=256, DT_RANK=8
// ---------------------------------------------------------------------------

#define BB 8
#define LL 4096
#define CC 256
#define DM 128
#define DS 16
#define DI 256
#define DTR 8
#define MROWS (BB * LL)   // 32768

typedef __attribute__((ext_vector_type(16))) _Float16 v16h;
typedef __attribute__((ext_vector_type(8)))  _Float16 h16x8;
typedef __attribute__((ext_vector_type(8)))  float    f32x8;

__device__ __forceinline__ float silu_f(float x) {
    return x / (1.0f + __expf(-x));
}
__device__ __forceinline__ float sigmoid_f(float x) {
    return 1.0f / (1.0f + __expf(-x));
}

// generic -> LDS byte offset (ptrtoint of an addrspace(3) pointer)
__device__ __forceinline__ unsigned lds_offset_of(const void* p) {
    return (unsigned)(unsigned long long)(__attribute__((address_space(3))) void*)(void*)p;
}

// ---------------------------------------------------------------------------
// WMMA GEMM:  C[M,N] = act( A[M,K] @ W[K,N] + bias )
// block = 256 threads = 8 waves (2 M-waves x 4 N-waves), block tile 64x128.
// Each wave owns a 32x32 output tile = 2x2 WMMA subtiles -> 4 v_wmma / K-step.
// K stepped by 32, global->register prefetch pipelined against WMMA compute.
// LDS layouts (f16): A [m][k] pitch 40, B transposed [n][k] pitch 40, so each
// lane fragment is two contiguous 16-byte ds_load_b128 per ISA layouts:
//   A 16x32: elem e<8 -> K=8g+e ; e>=8 -> K=16+8g+(e-8)   (g = lane/16)
//   B 32x16: elem e<8 -> K=8g+e ; e>=8 -> K=16+8g+(e-8) ; N = lane%16
// ---------------------------------------------------------------------------
#define MBLK 64
#define NBLK 128
#define KBLK 32
#define APITCH 40   // f16 elements; 80B rows keep fragment loads 16B aligned
#define BPITCH 40

__device__ __forceinline__ void stage_global_load(
    const float* __restrict__ A, const float* __restrict__ W,
    int N, int K, int kk, int m0, int n0,
    int ar, int ac, int bk, int bn, bool bfull,
    float (&a_reg)[8], float (&b_reg)[16])
{
    const float* asrc = A + (size_t)(m0 + ar) * K + kk + ac;
    float4 a0 = *(const float4*)(asrc);
    float4 a1 = *(const float4*)(asrc + 4);
    a_reg[0] = a0.x; a_reg[1] = a0.y; a_reg[2] = a0.z; a_reg[3] = a0.w;
    a_reg[4] = a1.x; a_reg[5] = a1.y; a_reg[6] = a1.z; a_reg[7] = a1.w;
    if (kk + 2 * KBLK < K)
        __builtin_prefetch(asrc + KBLK, 0, 1);

    if (bfull) {
        const float* wrow = W + (size_t)(kk + bk) * N + n0 + bn;
        #pragma unroll
        for (int i = 0; i < 4; ++i) {
            float4 v = *(const float4*)(wrow + 4 * i);
            b_reg[4 * i + 0] = v.x; b_reg[4 * i + 1] = v.y;
            b_reg[4 * i + 2] = v.z; b_reg[4 * i + 3] = v.w;
        }
    } else {
        const float* wrow = W + (size_t)(kk + bk) * N;
        #pragma unroll
        for (int i = 0; i < 16; ++i) {
            int n = n0 + bn + i;
            b_reg[i] = (n < N) ? wrow[n] : 0.0f;
        }
    }
}

__device__ __forceinline__ void stage_lds_store(
    _Float16* Alds, _Float16* Blds,
    int ar, int ac, int bk, int bn,
    const float (&a_reg)[8], const float (&b_reg)[16])
{
    h16x8 ah;
    #pragma unroll
    for (int i = 0; i < 8; ++i) ah[i] = (_Float16)a_reg[i];
    *(h16x8*)&Alds[ar * APITCH + ac] = ah;
    #pragma unroll
    for (int i = 0; i < 16; ++i)
        Blds[(bn + i) * BPITCH + bk] = (_Float16)b_reg[i];
}

__global__ __launch_bounds__(256)
void wmma_gemm_kernel(const float* __restrict__ A,
                      const float* __restrict__ W,
                      const float* __restrict__ bias,
                      float* __restrict__ C,
                      int M, int N, int K, int act)
{
    __shared__ __align__(16) _Float16 Alds[MBLK * APITCH];
    __shared__ __align__(16) _Float16 Blds[NBLK * BPITCH];   // transposed [n][k]

    const int tid  = threadIdx.x;
    const int lane = tid & 31;
    const int wave = tid >> 5;       // 0..7
    const int wm   = wave & 1;       // 2 waves along M
    const int wn   = wave >> 1;      // 4 waves along N
    const int m0   = blockIdx.x * MBLK;
    const int n0   = blockIdx.y * NBLK;

    f32x8 acc[2][2] = {};

    // staging assignments
    const int ar = tid >> 2;          // A: 64 rows, 4 threads/row
    const int ac = (tid & 3) * 8;     //    8 consecutive k each
    const int bk = tid >> 3;          // B: 32 k-rows, 8 threads/row
    const int bn = (tid & 7) * 16;    //    16 n each

    const bool bfull = (n0 + NBLK) <= N;

    float a_reg[8];
    float b_reg[16];

    // prologue: stage first K tile
    stage_global_load(A, W, N, K, 0, m0, n0, ar, ac, bk, bn, bfull, a_reg, b_reg);
    stage_lds_store(Alds, Blds, ar, ac, bk, bn, a_reg, b_reg);
    __syncthreads();

    const int g  = lane >> 4;     // 0/1
    const int ml = lane & 15;

    for (int kk = 0; kk < K; kk += KBLK) {
        const bool more = (kk + KBLK) < K;
        // prefetch next tile into registers while WMMAs run off LDS
        if (more)
            stage_global_load(A, W, N, K, kk + KBLK, m0, n0,
                              ar, ac, bk, bn, bfull, a_reg, b_reg);

        union { v16h v; h16x8 h[2]; } af[2], bfr[2];
        #pragma unroll
        for (int s = 0; s < 2; ++s) {
            const _Float16* arow = &Alds[(wm * 32 + s * 16 + ml) * APITCH];
            af[s].h[0] = *(const h16x8*)(arow + 8 * g);        // K = 8g + 0..7
            af[s].h[1] = *(const h16x8*)(arow + 16 + 8 * g);   // K = 16+8g+0..7
            const _Float16* brow = &Blds[(wn * 32 + s * 16 + ml) * BPITCH];
            bfr[s].h[0] = *(const h16x8*)(brow + 8 * g);       // K = 8g + 0..7
            bfr[s].h[1] = *(const h16x8*)(brow + 16 + 8 * g);  // K = 16+8g+0..7
        }
        #pragma unroll
        for (int i = 0; i < 2; ++i)
            #pragma unroll
            for (int j = 0; j < 2; ++j)
                acc[i][j] = __builtin_amdgcn_wmma_f32_16x16x32_f16(
                    false, af[i].v, false, bfr[j].v, (short)0,
                    acc[i][j], false, false);

        __syncthreads();
        if (more)
            stage_lds_store(Alds, Blds, ar, ac, bk, bn, a_reg, b_reg);
        __syncthreads();
    }

    // store D: row = m0 + wm*32 + i*16 + 8g + v ; col = n0 + wn*32 + j*16 + ml
    #pragma unroll
    for (int j = 0; j < 2; ++j) {
        const int col = n0 + wn * 32 + j * 16 + ml;
        if (col < N) {
            const float bv = bias ? bias[col] : 0.0f;
            #pragma unroll
            for (int i = 0; i < 2; ++i) {
                #pragma unroll
                for (int v = 0; v < 8; ++v) {
                    int row = m0 + wm * 32 + i * 16 + 8 * g + v;
                    float x = acc[i][j][v] + bv;
                    if (act == 1) x = silu_f(x);
                    C[(size_t)row * N + col] = x;
                }
            }
        }
    }
}

// ---------------------------------------------------------------------------
// Causal depthwise conv (D_CONV=4) over xi = xz[:, :, 0:256], fused SiLU.
// ---------------------------------------------------------------------------
__global__ __launch_bounds__(256)
void conv_silu_kernel(const float* __restrict__ xz,
                      const float* __restrict__ conv_w,
                      const float* __restrict__ conv_b,
                      float* __restrict__ xc)
{
    size_t idx = (size_t)blockIdx.x * blockDim.x + threadIdx.x;
    if (idx >= (size_t)MROWS * DI) return;
    int d = (int)(idx & (DI - 1));
    size_t row = idx >> 8;               // b*L + t
    int t = (int)(row & (LL - 1));
    float s = conv_b[d];
    #pragma unroll
    for (int k = 0; k < 4; ++k) {
        int tt = t - 3 + k;
        if (tt >= 0)
            s += conv_w[d * 4 + k] * xz[(row + (size_t)(tt - t)) * (2 * DI) + d];
    }
    xc[row * DI + d] = silu_f(s);
}

// ---------------------------------------------------------------------------
// delta = softplus(dt @ W_dt + b_dt), dt = dbc[:, 0:8]   (K=8 -> scalar FMAs)
// ---------------------------------------------------------------------------
__global__ __launch_bounds__(256)
void delta_kernel(const float* __restrict__ dbc,
                  const float* __restrict__ W_dt,
                  const float* __restrict__ b_dt,
                  float* __restrict__ dlt)
{
    size_t idx = (size_t)blockIdx.x * blockDim.x + threadIdx.x;
    if (idx >= (size_t)MROWS * DI) return;
    int d = (int)(idx & (DI - 1));
    size_t row = idx >> 8;
    const float* dt = dbc + row * (DTR + 2 * DS);
    float x = b_dt[d];
    #pragma unroll
    for (int j = 0; j < DTR; ++j) x += dt[j] * W_dt[j * DI + d];
    float sp = (x > 20.0f) ? x : log1pf(__expf(x));
    dlt[row * DI + d] = sp;
}

// ---------------------------------------------------------------------------
// Selective scan. One block per batch (256 threads = channels), h[16] in regs.
// B_t/C_t chunks staged with CDNA5 async global->LDS DMA (ASYNCcnt tracked):
// dbc[:, 8:40] rows are contiguous, so each thread issues two
// global_load_async_to_lds_b128 (inst offset applies to BOTH addresses).
// Output fused: yb = (ys + xc*D) * silu(z)
// ---------------------------------------------------------------------------
__global__ __launch_bounds__(256)
void scan_kernel(const float* __restrict__ dbc,
                 const float* __restrict__ dlt,
                 const float* __restrict__ xc,
                 const float* __restrict__ xz,
                 const float* __restrict__ A_log,
                 const float* __restrict__ D_param,
                 float* __restrict__ yb)
{
    __shared__ __align__(16) float BC[64][32];   // [t][0:16]=B_t, [t][16:32]=C_t

    const int b = blockIdx.x;
    const int d = threadIdx.x;

    float Ad[DS];
    #pragma unroll
    for (int n = 0; n < DS; ++n) Ad[n] = -__expf(A_log[d * DS + n]);
    const float Dp = D_param[d];

    float h[DS];
    #pragma unroll
    for (int n = 0; n < DS; ++n) h[n] = 0.0f;

    const int sr = threadIdx.x >> 2;        // 64 rows, 4 threads/row
    const int sc = (threadIdx.x & 3) * 8;   // 8 of 32 floats each
    const unsigned lds_dst = lds_offset_of(&BC[sr][sc]);

    for (int t0 = 0; t0 < LL; t0 += 64) {
        // async DMA: 32 bytes per thread, global -> LDS, no VGPR round-trip
        {
            const float* src = dbc + (size_t)(b * LL + t0 + sr) * (DTR + 2 * DS) + DTR + sc;
            asm volatile(
                "global_load_async_to_lds_b128 %0, %1, off\n\t"
                "global_load_async_to_lds_b128 %0, %1, off offset:16"
                :: "v"(lds_dst), "v"(src) : "memory");
        }
        asm volatile("s_wait_asynccnt 0x0" ::: "memory");
        __syncthreads();

        for (int tt = 0; tt < 64; ++tt) {
            size_t row = (size_t)b * LL + t0 + tt;
            float delta = dlt[row * DI + d];
            float u     = xc[row * DI + d];
            float du    = delta * u;
            float y     = 0.0f;
            #pragma unroll
            for (int n = 0; n < DS; ++n) {
                float e = __expf(delta * Ad[n]);
                h[n] = h[n] * e + du * BC[tt][n];
                y += h[n] * BC[tt][16 + n];
            }
            float z = xz[row * (2 * DI) + DI + d];
            yb[row * DI + d] = (y + u * Dp) * silu_f(z);
        }
        __syncthreads();
    }
}

// ---------------------------------------------------------------------------
// Gating head. One block per row; 16-wide hidden vectors via LDS.
// out = (mf@W_out + b_out) * sigmoid(texture) * (1 - sigmoid(semantic))^2
// ---------------------------------------------------------------------------
__global__ __launch_bounds__(256)
void gate_kernel(const float* __restrict__ outpre,
                 const float* __restrict__ semh,
                 const float* __restrict__ freq,
                 const float* __restrict__ Wt1, const float* __restrict__ bt1,
                 const float* __restrict__ Wt2, const float* __restrict__ bt2,
                 const float* __restrict__ Ws2, const float* __restrict__ bs2,
                 float* __restrict__ out)
{
    __shared__ float th[16];
    __shared__ float sh[16];
    size_t row = blockIdx.x;
    int c = threadIdx.x;
    if (c < 16) {
        float f = freq[row];
        th[c] = silu_f(f * Wt1[c] + bt1[c]);
        sh[c] = semh[row * 16 + c];
    }
    __syncthreads();
    float st = bt2[c];
    float im = bs2[c];
    #pragma unroll
    for (int j = 0; j < 16; ++j) {
        st += th[j] * Wt2[j * CC + c];
        im += sh[j] * Ws2[j * CC + c];
    }
    st = sigmoid_f(st);
    im = sigmoid_f(im);
    float g = st * (1.0f - im) * (1.0f - im);
    out[row * CC + c] = outpre[row * CC + c] * g;
}

// ---------------------------------------------------------------------------
// Launch pipeline
// ---------------------------------------------------------------------------
extern "C" void kernel_launch(void* const* d_in, const int* in_sizes, int n_in,
                              void* d_out, int out_size, void* d_ws, size_t ws_size,
                              hipStream_t stream)
{
    (void)in_sizes; (void)n_in; (void)out_size; (void)ws_size;

    const float* x        = (const float*)d_in[0];
    const float* freq     = (const float*)d_in[1];
    const float* W_in     = (const float*)d_in[2];
    const float* b_in     = (const float*)d_in[3];
    const float* W_inproj = (const float*)d_in[4];
    const float* conv_w   = (const float*)d_in[5];
    const float* conv_b   = (const float*)d_in[6];
    const float* W_xproj  = (const float*)d_in[7];
    const float* W_dt     = (const float*)d_in[8];
    const float* b_dt     = (const float*)d_in[9];
    const float* A_log    = (const float*)d_in[10];
    const float* D_param  = (const float*)d_in[11];
    const float* W_outprj = (const float*)d_in[12];
    const float* Wt1      = (const float*)d_in[13];
    const float* bt1      = (const float*)d_in[14];
    const float* Wt2      = (const float*)d_in[15];
    const float* bt2      = (const float*)d_in[16];
    const float* Ws1      = (const float*)d_in[17];
    const float* bs1      = (const float*)d_in[18];
    const float* Ws2      = (const float*)d_in[19];
    const float* bs2      = (const float*)d_in[20];
    const float* W_out    = (const float*)d_in[21];
    const float* b_out    = (const float*)d_in[22];
    float*       out      = (float*)d_out;

    // workspace layout (fp32 elements), ~190 MB peak with reuse
    float* ws   = (float*)d_ws;
    float* feat = ws;                              // M*128
    float* xz   = feat + (size_t)MROWS * DM;       // M*512
    float* xc   = xz   + (size_t)MROWS * 2 * DI;   // M*256
    float* dbc  = xc   + (size_t)MROWS * DI;       // M*40
    float* dlt  = dbc  + (size_t)MROWS * (DTR + 2 * DS); // M*256
    float* yb   = dlt  + (size_t)MROWS * DI;       // M*256
    float* mf     = feat;   // reuse: feat dead after GEMM2
    float* semh   = dbc;    // reuse: dbc dead after scan (M*16 <= M*40)
    float* outpre = xc;     // reuse: xc dead after scan

    const int M = MROWS;
    dim3 blk(256);

    // 1) feat = x @ W_in + b_in                      [K=256 -> N=128]
    wmma_gemm_kernel<<<dim3(M / MBLK, 1), blk, 0, stream>>>(
        x, W_in, b_in, feat, M, DM, CC, 0);
    // 2) xz = feat @ W_inproj                        [K=128 -> N=512]
    wmma_gemm_kernel<<<dim3(M / MBLK, (2 * DI) / NBLK), blk, 0, stream>>>(
        feat, W_inproj, nullptr, xz, M, 2 * DI, DM, 0);
    // 3) xc = silu(causal_conv(xi))
    conv_silu_kernel<<<dim3((M * DI) / 256), blk, 0, stream>>>(xz, conv_w, conv_b, xc);
    // 4) dbc = xc @ W_xproj                          [K=256 -> N=40]
    wmma_gemm_kernel<<<dim3(M / MBLK, 1), blk, 0, stream>>>(
        xc, W_xproj, nullptr, dbc, M, DTR + 2 * DS, DI, 0);
    // 5) delta = softplus(dt @ W_dt + b_dt)
    delta_kernel<<<dim3((M * DI) / 256), blk, 0, stream>>>(dbc, W_dt, b_dt, dlt);
    // 6) selective scan (fused (ys + xc*D) * silu(z)), async-to-LDS staging
    scan_kernel<<<dim3(BB), blk, 0, stream>>>(dbc, dlt, xc, xz, A_log, D_param, yb);
    // 7) mf = yb @ W_outproj                         [K=256 -> N=128]
    wmma_gemm_kernel<<<dim3(M / MBLK, 1), blk, 0, stream>>>(
        yb, W_outprj, nullptr, mf, M, DM, DI, 0);
    // 8) semh = silu(mf @ Ws1 + bs1)                 [K=128 -> N=16]
    wmma_gemm_kernel<<<dim3(M / MBLK, 1), blk, 0, stream>>>(
        mf, Ws1, bs1, semh, M, 16, DM, 1);
    // 9) outpre = mf @ W_out + b_out                 [K=128 -> N=256]
    wmma_gemm_kernel<<<dim3(M / MBLK, CC / NBLK), blk, 0, stream>>>(
        mf, W_out, b_out, outpre, M, CC, DM, 0);
    // 10) gating
    gate_kernel<<<dim3(M), blk, 0, stream>>>(
        outpre, semh, freq, Wt1, bt1, Wt2, bt2, Ws2, bs2, out);
}
